// ProbProtoSegHead_37065567765238
// MI455X (gfx1250) — matrix operand order repaired and checked by reading, not compiled, not run on hardware
//
#include <hip/hip_runtime.h>
#include <hip/hip_bf16.h>

// Problem constants (from reference): N=131072 pixels, K=256, C*M=190 prototypes, R=2.
#define NPIX   131072
#define KDIM   256
#define CM     190
#define CMPAD  192          // 12 tiles of 16
#define KB     512          // concatenated K for the fused GEMM (xbar | sqrt(var))
#define NT     (NPIX / 16)  // 8192 n-tiles

typedef __attribute__((ext_vector_type(16))) _Float16 v16h;
typedef __attribute__((ext_vector_type(8)))  _Float16 v8h;
typedef __attribute__((ext_vector_type(8)))  float    v8f;

__device__ __forceinline__ float wred32(float v) {
#pragma unroll
    for (int m = 16; m > 0; m >>= 1) v += __shfl_xor(v, m, 32);
    return v;
}

__device__ __forceinline__ void load8(const float* __restrict__ p, float* d) {
    float4 a = ((const float4*)p)[0];
    float4 b = ((const float4*)p)[1];
    d[0] = a.x; d[1] = a.y; d[2] = a.z; d[3] = a.w;
    d[4] = b.x; d[5] = b.y; d[6] = b.z; d[7] = b.w;
}

// Scatter one 8-half segment of logical row n at column K0 (multiple of 8, 0..511)
// into the WMMA *A-fragment-native* layout (ISA 7.12.2, 16-bit A 16x32):
//   fragment index = nt*16 + kc  (kc = K0/32), 32 lanes x 16 halves per fragment.
//   lanes 0-15: M=lane, K {0..7, 16..23}; lanes 16-31: M=lane-16, K {8..15, 24..31}.
__device__ __forceinline__ void write_segA(_Float16* __restrict__ A, int n, int K0,
                                           const float* v, float scale) {
    int kc = K0 >> 5, r = K0 & 31;            // r in {0,8,16,24}
    int m = n & 15, nt = n >> 4;
    int fragLane = m + ((r & 8) ? 16 : 0);
    int halfOff  = (r & 16) ? 8 : 0;
    size_t off = (((size_t)nt * 16 + kc) * 32 + fragLane) * 16 + halfOff;
    v8h s;
#pragma unroll
    for (int j = 0; j < 8; j++) s[j] = (_Float16)(v[j] * scale);
    *(v8h*)(A + off) = s;
}

// Same for the B fragment layout (16-bit B 32x16):
//   lanes 0-15 hold K {0..7, 16..23}, lanes 16-31 hold K {8..15, 24..31}, N = lane%16.
//   fragment index = kc*12 + t  (t = cm/16).
__device__ __forceinline__ void write_segB(_Float16* __restrict__ B, int cm, int K0,
                                           const float* v, float scale) {
    int kc = K0 >> 5, r = K0 & 31;
    int nl = cm & 15, t = cm >> 4;
    int fragLane = nl + ((r & 8) ? 16 : 0);
    int halfOff  = (r & 16) ? 8 : 0;
    size_t off = (((size_t)kc * 12 + t) * 32 + fragLane) * 16 + halfOff;
    v8h s;
#pragma unroll
    for (int j = 0; j < 8; j++) s[j] = (_Float16)(v[j] * scale);
    *(v8h*)(B + off) = s;
}

// ---------------- Pass 1: per-pixel reparam + LN + L2, averaged over R=2 ----------------
// One wave32 per row; lane owns 8 consecutive columns. Writes A (f16, frag-swizzled)
// and bias_x[n] = -2 - (1/K)*sum(x_var[n]).
__global__ void __launch_bounds__(256) prep_x_kernel(
    const float* __restrict__ x, const float* __restrict__ xv,
    const float* __restrict__ gamma, const float* __restrict__ beta,
    const float* __restrict__ eps, _Float16* __restrict__ A,
    float* __restrict__ bias_x)
{
    const int lane = threadIdx.x & 31;
    const int n = blockIdx.x * 8 + (threadIdx.x >> 5);
    const int c0 = lane * 8;
    const size_t rb = (size_t)n * KDIM + c0;

    float xr[8], vr[8], g8[8], b8[8];
    load8(x + rb, xr);
    load8(xv + rb, vr);
    load8(gamma + c0, g8);
    load8(beta + c0, b8);

    float acc8[8];
#pragma unroll
    for (int j = 0; j < 8; j++) acc8[j] = 0.f;

#pragma unroll
    for (int r = 0; r < 2; r++) {
        float e[8];
        load8(eps + (size_t)r * NPIX * KDIM + rb, e);
        float v[8], s1 = 0.f, s2 = 0.f;
#pragma unroll
        for (int j = 0; j < 8; j++) {
            v[j] = fmaf(e[j], vr[j], xr[j]);
            s1 += v[j];
            s2 += v[j] * v[j];
        }
        s1 = wred32(s1); s2 = wred32(s2);
        float mu  = s1 * (1.f / KDIM);
        float var = s2 * (1.f / KDIM) - mu * mu;
        float inv = rsqrtf(var + 1e-5f);
        float y[8], q = 0.f;
#pragma unroll
        for (int j = 0; j < 8; j++) {
            y[j] = fmaf((v[j] - mu) * inv, g8[j], b8[j]);
            q += y[j] * y[j];
        }
        q = wred32(q);
        float rn = 1.f / fmaxf(sqrtf(q), 1e-12f);
#pragma unroll
        for (int j = 0; j < 8; j++) acc8[j] += 0.5f * y[j] * rn;   // mean over R=2
    }

    float sxv = 0.f;
#pragma unroll
    for (int j = 0; j < 8; j++) sxv += vr[j];
    sxv = wred32(sxv);
    if (lane == 0) bias_x[n] = -2.f - sxv * (1.f / KDIM);

    float sq[8];
#pragma unroll
    for (int j = 0; j < 8; j++) sq[j] = sqrtf(vr[j]);

    write_segA(A, n, c0, acc8, 1.f);          // xbar  -> K 0..255
    write_segA(A, n, KDIM + c0, sq, 1.f);     // sqrt(x_var) -> K 256..511
}

// ---------------- Pass 2: prototypes (190 rows, padded to 192 zero rows) ----------------
__global__ void __launch_bounds__(256) prep_p_kernel(
    const float* __restrict__ p, const float* __restrict__ pv,
    const float* __restrict__ gamma, const float* __restrict__ beta,
    const float* __restrict__ eps, _Float16* __restrict__ B,
    float* __restrict__ bias_p)
{
    const int lane = threadIdx.x & 31;
    const int cm = blockIdx.x * 8 + (threadIdx.x >> 5);
    const int c0 = lane * 8;

    if (cm >= CM) {   // zero padding rows 190,191
        float z[8] = {0.f, 0.f, 0.f, 0.f, 0.f, 0.f, 0.f, 0.f};
        write_segB(B, cm, c0, z, 1.f);
        write_segB(B, cm, KDIM + c0, z, 1.f);
        if (lane == 0) bias_p[cm] = 0.f;
        return;
    }

    const size_t rb = (size_t)cm * KDIM + c0;
    float pr[8], vr[8], g8[8], b8[8];
    load8(p + rb, pr);
    load8(pv + rb, vr);
    load8(gamma + c0, g8);
    load8(beta + c0, b8);

    float acc8[8];
#pragma unroll
    for (int j = 0; j < 8; j++) acc8[j] = 0.f;

#pragma unroll
    for (int r = 0; r < 2; r++) {
        float e[8];
        load8(eps + (size_t)r * CM * KDIM + rb, e);
        float v[8], s1 = 0.f, s2 = 0.f;
#pragma unroll
        for (int j = 0; j < 8; j++) {
            v[j] = fmaf(e[j], vr[j], pr[j]);
            s1 += v[j];
            s2 += v[j] * v[j];
        }
        s1 = wred32(s1); s2 = wred32(s2);
        float mu  = s1 * (1.f / KDIM);
        float var = s2 * (1.f / KDIM) - mu * mu;
        float inv = rsqrtf(var + 1e-5f);
        float y[8], q = 0.f;
#pragma unroll
        for (int j = 0; j < 8; j++) {
            y[j] = fmaf((v[j] - mu) * inv, g8[j], b8[j]);
            q += y[j] * y[j];
        }
        q = wred32(q);
        float rn = 1.f / fmaxf(sqrtf(q), 1e-12f);
#pragma unroll
        for (int j = 0; j < 8; j++) acc8[j] += 0.5f * y[j] * rn;
    }

    float spv = 0.f;
#pragma unroll
    for (int j = 0; j < 8; j++) spv += vr[j];
    spv = wred32(spv);
    if (lane == 0) bias_p[cm] = -spv * (1.f / KDIM);

    float sq[8];
#pragma unroll
    for (int j = 0; j < 8; j++) sq[j] = sqrtf(vr[j]);

    write_segB(B, cm, c0, acc8, 2.f);                    // 2*pbar
    write_segB(B, cm, KDIM + c0, sq, 2.f / KDIM);        // (2/K)*sqrt(p_var)
}

// ---------------- Pass 3: fused WMMA GEMM  out[N,190] = A[N,512] x B[192,512]^T + biases
// One wave32 per 16-row n-tile; it owns all 12 cm-tiles (12 x v8f accumulators).
// A and B are already in WMMA-fragment-native layout: each lane's operand is one
// contiguous 32B v16h load.
__global__ void __launch_bounds__(256) gemm_kernel(
    const _Float16* __restrict__ A, const _Float16* __restrict__ B,
    const float* __restrict__ bias_x, const float* __restrict__ bias_p,
    float* __restrict__ out)
{
    const int lane = threadIdx.x & 31;
    const int nt = blockIdx.x * 8 + (threadIdx.x >> 5);

    v8f acc[12];
    v8f z = {};
#pragma unroll
    for (int t = 0; t < 12; t++) acc[t] = z;

    for (int kc = 0; kc < 16; kc++) {   // 16 chunks of K=32 over KB=512
        v16h a = *(const v16h*)(A + (((size_t)nt * 16 + kc) * 32 + lane) * 16);
#pragma unroll
        for (int t = 0; t < 12; t++) {
            v16h b = *(const v16h*)(B + (((size_t)kc * 12 + t) * 32 + lane) * 16);
            acc[t] = __builtin_amdgcn_wmma_f32_16x16x32_f16(
                false, a, false, b, (short)0, acc[t], false, false);
        }
    }

    // C/D layout (ISA 7.12.2): VGPR j, lanes 0-15 -> M=j, lanes 16-31 -> M=j+8; N=lane%16.
    const int colBase = lane & 15;
    const int rowOff  = (lane & 16) ? 8 : 0;
    float bx[8];
#pragma unroll
    for (int j = 0; j < 8; j++) bx[j] = bias_x[nt * 16 + rowOff + j];

#pragma unroll
    for (int t = 0; t < 12; t++) {
        int col = t * 16 + colBase;
        if (col < CM) {
            float bp = bias_p[col];
#pragma unroll
            for (int j = 0; j < 8; j++) {
                size_t n = (size_t)nt * 16 + rowOff + j;
                out[n * CM + col] = acc[t][j] + bx[j] + bp;
            }
        }
    }
}

extern "C" void kernel_launch(void* const* d_in, const int* in_sizes, int n_in,
                              void* d_out, int out_size, void* d_ws, size_t ws_size,
                              hipStream_t stream) {
    const float* x      = (const float*)d_in[0];
    const float* x_var  = (const float*)d_in[1];
    const float* proto  = (const float*)d_in[2];
    const float* pvar   = (const float*)d_in[3];
    const float* gamma  = (const float*)d_in[4];
    const float* beta   = (const float*)d_in[5];
    const float* eps_x  = (const float*)d_in[6];
    const float* eps_p  = (const float*)d_in[7];
    float* out = (float*)d_out;

    char* ws = (char*)d_ws;
    _Float16* A = (_Float16*)ws;                                   // 131072*512 f16 = 128 MB
    _Float16* B = (_Float16*)(ws + (size_t)NPIX * KB * 2);         // 192*512 f16 = 192 KB
    float* bias_x = (float*)(ws + (size_t)NPIX * KB * 2 + (size_t)CMPAD * KB * 2);
    float* bias_p = bias_x + NPIX;

    prep_x_kernel<<<NPIX / 8, 256, 0, stream>>>(x, x_var, gamma, beta, eps_x, A, bias_x);
    prep_p_kernel<<<CMPAD / 8, 256, 0, stream>>>(proto, pvar, gamma, beta, eps_p, B, bias_p);
    gemm_kernel<<<NT / 8, 256, 0, stream>>>(A, B, bias_x, bias_p, out);
}
